// FaceImageClassifier_69526930587890
// MI455X (gfx1250) — matrix operand order repaired and checked by reading, not compiled, not run on hardware
//
#include <hip/hip_runtime.h>
#include <hip/hip_bf16.h>
#include <math.h>
#include <stdint.h>

// ---------------- types ----------------
typedef __attribute__((ext_vector_type(16))) __bf16 v16bf;
typedef __attribute__((ext_vector_type(8)))  __bf16 v8bf;
typedef __attribute__((ext_vector_type(8)))  float  v8f;

// ---------------- problem dims ----------------
static constexpr int kB = 256;    // batch
static constexpr int kT = 128;    // timesteps
static constexpr int kH = 512;    // hidden (== input)
static constexpr int kG = 2048;   // 4*H gates
static constexpr int kL = 5;      // layers
static constexpr int kC = 40;     // classes
static constexpr int kM = kB * kT;  // 32768 rows of batched GEMM

// ---------------- helpers ----------------
__device__ __forceinline__ v16bf make_v16(v8bf lo, v8bf hi) {
    v16bf r;
#pragma unroll
    for (int e = 0; e < 8; ++e) { r[e] = lo[e]; r[8 + e] = hi[e]; }
    return r;
}

__device__ __forceinline__ float sigmoidf_fast(float x) {
    return 1.0f / (1.0f + __expf(-x));
}

// CDNA5 async global->LDS copy, 16 bytes per lane, tracked by ASYNCcnt.
// vdst VGPR carries the per-lane LDS byte address (low 32 bits of the
// generic pointer == LDS aperture offset per the CDNA5 address map).
__device__ __forceinline__ void async_copy_b128(uint32_t lds_byte_addr, const void* gaddr) {
    asm volatile("global_load_async_to_lds_b128 %0, %1, off"
                 :: "v"(lds_byte_addr), "v"((uint64_t)(uintptr_t)gaddr)
                 : "memory");
}
__device__ __forceinline__ void wait_asynccnt0() {
    asm volatile("s_wait_asynccnt 0x0" ::: "memory");
}

// ---------------- f32 -> bf16 convert ----------------
__global__ void cvt_f32_bf16(const float* __restrict__ s, __bf16* __restrict__ d, int n) {
    int i = blockIdx.x * blockDim.x + threadIdx.x;
    int stride = gridDim.x * blockDim.x;
    for (; i < n; i += stride) d[i] = (__bf16)s[i];
}

// ---------------- batched input-projection GEMM ----------------
// XW[m][n] = sum_k X[m][k] * W[n][k] + bih[n] + bhh[n]
// M = 32768, N = 2048, K = 512.  Block: 128x128 tile, 256 threads (8 waves),
// wave w owns a 16(M) x 128(N) strip.  LDS tiles are double-buffered and
// filled with global_load_async_to_lds_b128 (ASYNCcnt), one barrier / chunk.
__global__ __launch_bounds__(256)
void gates_gemm(const __bf16* __restrict__ X, const __bf16* __restrict__ W,
                const float* __restrict__ bih, const float* __restrict__ bhh,
                float* __restrict__ XW) {
    __shared__ __bf16 As[2][128 * 40];   // 128 rows x 32 K, stride 40 (pad, 16B aligned)
    __shared__ __bf16 Bs[2][128 * 40];

    const int tid  = threadIdx.x;
    const int wave = tid >> 5;
    const int lane = tid & 31;
    const int half = lane >> 4;
    const int r    = lane & 15;
    const int m0   = blockIdx.y * 128;
    const int n0   = blockIdx.x * 128;
    const int lrow = tid >> 1;          // staging: row 0..127
    const int lcol = (tid & 1) * 16;    // staging: 16 halves each

    // per-thread global staging sources (k advances by kk)
    const __bf16* xg0 = X + (size_t)(m0 + lrow) * kH + lcol;
    const __bf16* wg0 = W + (size_t)(n0 + lrow) * kH + lcol;
    // per-thread LDS staging destinations (byte addresses)
    uint32_t ldsA[2], ldsB[2];
#pragma unroll
    for (int c = 0; c < 2; ++c) {
        ldsA[c] = (uint32_t)(uintptr_t)&As[c][lrow * 40 + lcol];
        ldsB[c] = (uint32_t)(uintptr_t)&Bs[c][lrow * 40 + lcol];
    }

    v8f acc[8];
#pragma unroll
    for (int i = 0; i < 8; ++i)
#pragma unroll
        for (int j = 0; j < 8; ++j) acc[i][j] = 0.0f;

    // prefetch first K-chunk into buffer 0
    async_copy_b128(ldsA[0],      xg0);
    async_copy_b128(ldsA[0] + 16, xg0 + 8);
    async_copy_b128(ldsB[0],      wg0);
    async_copy_b128(ldsB[0] + 16, wg0 + 8);

    int cur = 0;
    for (int kk = 0; kk < kH; kk += 32) {
        wait_asynccnt0();      // my async fills of buffer `cur` are in LDS
        __syncthreads();       // everyone's fills done; prior reads retired

        if (kk + 32 < kH) {    // prefetch next chunk into the other buffer
            const int nxt = cur ^ 1;
            async_copy_b128(ldsA[nxt],      xg0 + kk + 32);
            async_copy_b128(ldsA[nxt] + 16, xg0 + kk + 40);
            async_copy_b128(ldsB[nxt],      wg0 + kk + 32);
            async_copy_b128(ldsB[nxt] + 16, wg0 + kk + 40);
        }

        // A fragment: row = wave*16 + r; elems 0..7 <-> K=half*8+0..7,
        //             elems 8..15 <-> K=16+half*8+0..7
        const __bf16* ap = &As[cur][(wave * 16 + r) * 40 + half * 8];
        v16bf a = make_v16(*(const v8bf*)ap, *(const v8bf*)(ap + 16));

        // preload all 8 B fragments into distinct regs, then WMMA burst
        v16bf bf[8];
#pragma unroll
        for (int nt = 0; nt < 8; ++nt) {
            const __bf16* bp = &Bs[cur][(nt * 16 + r) * 40 + half * 8];
            bf[nt] = make_v16(*(const v8bf*)bp, *(const v8bf*)(bp + 16));
        }
#pragma unroll
        for (int nt = 0; nt < 8; ++nt) {
            acc[nt] = __builtin_amdgcn_wmma_f32_16x16x32_bf16(
                false, a, false, bf[nt], (short)0, acc[nt], false, false);
        }
        cur ^= 1;
    }

#pragma unroll
    for (int nt = 0; nt < 8; ++nt) {
        const int col = n0 + nt * 16 + r;
        const float bias = bih[col] + bhh[col];
#pragma unroll
        for (int j = 0; j < 8; ++j) {
            const int row = m0 + wave * 16 + j + half * 8;   // C: M = j + 8*half, N = r
            XW[(size_t)row * kG + col] = acc[nt][j] + bias;
        }
    }
}

// ---------------- fused recurrent step ----------------
// G = XW[:,t,:] + h_{t-1} @ Whh^T ; then LSTM gate math, write c and h(bf16).
// Block: 32 batch rows x 64 h-cols. Wave (mt,jt) computes the same 16x16 (b,j)
// tile for ALL four gates -> elementwise update entirely in registers.
// Whh (2MB bf16) is L2-resident; fragments load straight from global.
__global__ __launch_bounds__(256)
void lstm_step(const float* __restrict__ XW, const __bf16* __restrict__ Whh,
               __bf16* __restrict__ act, float* __restrict__ cst, int t) {
    const int tid  = threadIdx.x;
    const int wave = tid >> 5;
    const int lane = tid & 31;
    const int half = lane >> 4;
    const int r    = lane & 15;
    const int mt   = wave >> 2;   // 0..1 : batch tile
    const int jt   = wave & 3;    // 0..3 : h-col tile
    const int b0   = blockIdx.y * 32;
    const int j0   = blockIdx.x * 64;

    v8f acc[4];
#pragma unroll
    for (int g = 0; g < 4; ++g)
#pragma unroll
        for (int j = 0; j < 8; ++j) acc[g][j] = 0.0f;

    if (t > 0) {
        // h_{t-1} lives in the activation buffer written at step t-1
        const __bf16* hrow = act + (size_t)(b0 + mt * 16 + r) * (kT * kH) + (size_t)(t - 1) * kH;
        for (int kk = 0; kk < kH; kk += 32) {
            const __bf16* ap = hrow + kk + half * 8;
            v16bf a = make_v16(*(const v8bf*)ap, *(const v8bf*)(ap + 16));
            v16bf bf[4];
#pragma unroll
            for (int g = 0; g < 4; ++g) {
                const __bf16* bp = Whh + (size_t)(g * kH + j0 + jt * 16 + r) * kH + kk + half * 8;
                bf[g] = make_v16(*(const v8bf*)bp, *(const v8bf*)(bp + 16));
            }
#pragma unroll
            for (int g = 0; g < 4; ++g) {
                acc[g] = __builtin_amdgcn_wmma_f32_16x16x32_bf16(
                    false, a, false, bf[g], (short)0, acc[g], false, false);
            }
        }
    }

    const int col = j0 + jt * 16 + r;
#pragma unroll
    for (int j = 0; j < 8; ++j) {
        const int row = b0 + mt * 16 + j + half * 8;              // batch index
        const size_t xwbase = ((size_t)row * kT + t) * kG;
        float gi = acc[0][j] + XW[xwbase + col];
        float gf = acc[1][j] + XW[xwbase + kH + col];
        float gg = acc[2][j] + XW[xwbase + 2 * kH + col];
        float go = acc[3][j] + XW[xwbase + 3 * kH + col];
        float cp = (t > 0) ? cst[(size_t)row * kH + col] : 0.0f;
        float cn = sigmoidf_fast(gf) * cp + sigmoidf_fast(gi) * tanhf(gg);
        float hn = sigmoidf_fast(go) * tanhf(cn);
        cst[(size_t)row * kH + col] = cn;
        act[(size_t)row * (kT * kH) + (size_t)t * kH + col] = (__bf16)hn;
    }
}

// ---------------- final FC (256x40x512 -- negligible) ----------------
__global__ void fc_kernel(const __bf16* __restrict__ act, const float* __restrict__ Wfc,
                          const float* __restrict__ bfc, float* __restrict__ out) {
    const int b = blockIdx.x;
    const int c = threadIdx.x;
    if (c >= kC) return;
    const __bf16* h = act + (size_t)b * (kT * kH) + (size_t)(kT - 1) * kH;
    float s = bfc[c];
    for (int j = 0; j < kH; ++j) s += (float)h[j] * Wfc[c * kH + j];
    out[b * kC + c] = s;
}

// ---------------- workspace layout (bytes) ----------------
static constexpr size_t OFF_WIHB = 0;                                   // 5*2048*512 bf16
static constexpr size_t SZ_WB    = (size_t)kL * kG * kH * 2;            // 10,485,760
static constexpr size_t OFF_WHHB = OFF_WIHB + SZ_WB;
static constexpr size_t OFF_ACTA = OFF_WHHB + SZ_WB;                    // B*T*H bf16
static constexpr size_t SZ_ACT   = (size_t)kB * kT * kH * 2;            // 33,554,432
static constexpr size_t OFF_ACTB = OFF_ACTA + SZ_ACT;
static constexpr size_t OFF_XW   = OFF_ACTB + SZ_ACT;                   // B*T*4H f32
static constexpr size_t SZ_XW    = (size_t)kB * kT * kG * 4;            // 268,435,456
static constexpr size_t OFF_CST  = OFF_XW + SZ_XW;                      // B*H f32

extern "C" void kernel_launch(void* const* d_in, const int* in_sizes, int n_in,
                              void* d_out, int out_size, void* d_ws, size_t ws_size,
                              hipStream_t stream) {
    const float* x    = (const float*)d_in[0];
    const float* Wih  = (const float*)d_in[1];
    const float* Whh  = (const float*)d_in[2];
    const float* bih  = (const float*)d_in[3];
    const float* bhh  = (const float*)d_in[4];
    const float* Wfc  = (const float*)d_in[5];
    const float* bfc  = (const float*)d_in[6];

    char* ws = (char*)d_ws;
    __bf16* WihB = (__bf16*)(ws + OFF_WIHB);
    __bf16* WhhB = (__bf16*)(ws + OFF_WHHB);
    __bf16* actA = (__bf16*)(ws + OFF_ACTA);
    __bf16* actB = (__bf16*)(ws + OFF_ACTB);
    float*  XW   = (float*)(ws + OFF_XW);
    float*  cst  = (float*)(ws + OFF_CST);

    const int nW = kL * kG * kH;          // 5,242,880
    cvt_f32_bf16<<<2048, 256, 0, stream>>>(Wih, WihB, nW);
    cvt_f32_bf16<<<2048, 256, 0, stream>>>(Whh, WhhB, nW);
    cvt_f32_bf16<<<4096, 256, 0, stream>>>(x, actA, kB * kT * kH);

    __bf16* bufs[2] = {actA, actB};
    for (int l = 0; l < kL; ++l) {
        __bf16* in  = bufs[l & 1];
        __bf16* out = bufs[(l + 1) & 1];
        dim3 gg(kG / 128, kM / 128);      // 16 x 256 blocks
        gates_gemm<<<gg, 256, 0, stream>>>(in, WihB + (size_t)l * kG * kH,
                                           bih + l * kG, bhh + l * kG, XW);
        for (int t = 0; t < kT; ++t) {
            lstm_step<<<dim3(kH / 64, kB / 32), 256, 0, stream>>>(
                XW, WhhB + (size_t)l * kG * kH, out, cst, t);
        }
    }
    fc_kernel<<<kB, 64, 0, stream>>>(bufs[kL & 1], Wfc, bfc, (float*)d_out);
}
